// EMAVectorQuantizer2_26439818674879
// MI455X (gfx1250) — matrix operand order repaired and checked by reading, not compiled, not run on hardware
//
#include <hip/hip_runtime.h>
#include <hip/hip_bf16.h>
#include <math.h>

// ---------------------------------------------------------------------------
// VQ (EMAVectorQuantizer) for MI455X / gfx1250, wave32, WMMA bf16 3-term GEMM,
// double-buffered async global->LDS staging of codebook tiles.
//
// z:   [16, 256, 32, 32] f32  (N = 16384 vectors of dim 256, NCHW)
// emb: [8192, 256] f32
// out (flat f32): z_q[4194304] | loss[1] | perplexity[1] |
//                 encodings[16384*8192] | encoding_indices[16384]
// ---------------------------------------------------------------------------

typedef __attribute__((ext_vector_type(16))) __bf16 v16bf;
typedef __attribute__((ext_vector_type(8)))  __bf16 v8bf;
typedef __attribute__((ext_vector_type(8)))  float  v8f;
typedef int v4i __attribute__((vector_size(16)));   // matches builtin V4i

#define DIM      256
#define NCODES   8192
#define NVEC     16384
#define NELEM    4194304          // 16*256*32*32

// d_out float offsets
#define OUT_LOSS  4194304u
#define OUT_PERP  4194305u
#define OUT_ENC   4194306ull
#define OUT_IDX   138412034ull

// workspace byte offsets (total ~25.3 MB)
#define WS_EMB_HI   0ull
#define WS_EMB_LO   4194304ull
#define WS_Z_HI     8388608ull
#define WS_Z_LO     16777216ull
#define WS_ENORM    25165824ull
#define WS_IDX      25198592ull
#define WS_HIST     25264128ull
#define WS_LACC     25296896ull

#define GLOBAL_AS __attribute__((address_space(1)))
#define LDS_AS    __attribute__((address_space(3)))

// --- async global->LDS (CDNA5 ASYNCcnt path), with safe fallback -----------
#if defined(__has_builtin)
#if __has_builtin(__builtin_amdgcn_global_load_async_to_lds_b128)
#define HAVE_ASYNC_LDS 1
#endif
#endif
#ifndef HAVE_ASYNC_LDS
#define HAVE_ASYNC_LDS 0
#endif

#if HAVE_ASYNC_LDS
#if __has_builtin(__builtin_amdgcn_s_wait_asynccnt)
#define WAIT_ASYNC() __builtin_amdgcn_s_wait_asynccnt(0)
#else
#define WAIT_ASYNC() asm volatile("s_wait_asynccnt 0x0" ::: "memory")
#endif
#else
#define WAIT_ASYNC() do {} while (0)
#endif

__device__ __forceinline__ unsigned short f2bf(float x) {
    unsigned u = __float_as_uint(x);
    return (unsigned short)((u + 0x7FFFu + ((u >> 16) & 1u)) >> 16);  // RNE
}
__device__ __forceinline__ float bf2f(unsigned short b) {
    return __uint_as_float(((unsigned)b) << 16);
}

// --------------------------------------------------------------------------
// emb -> bf16 hi/lo split + per-code squared norm.  One block per code row.
// --------------------------------------------------------------------------
__global__ void vq_emb_prep(const float* __restrict__ emb,
                            unsigned short* __restrict__ ehi,
                            unsigned short* __restrict__ elo,
                            float* __restrict__ enorm) {
    int k = blockIdx.x, c = threadIdx.x;
    size_t o = (size_t)k * DIM + c;
    float e = emb[o];
    unsigned short hb = f2bf(e);
    unsigned short lb = f2bf(e - bf2f(hb));
    ehi[o] = hb;
    elo[o] = lb;
    __shared__ float red[256];
    red[c] = e * e;
    __syncthreads();
    for (int s = 128; s > 0; s >>= 1) {
        if (c < s) red[c] += red[c + s];
        __syncthreads();
    }
    if (c == 0) enorm[k] = red[0];
}

// --------------------------------------------------------------------------
// z NCHW f32 -> zf [N][256] bf16 hi/lo via LDS tile (coalesced R + W).
// One block per (b,h): 32 w-positions x 256 channels.
// --------------------------------------------------------------------------
__global__ void vq_z_prep(const float* __restrict__ z,
                          unsigned short* __restrict__ zhi,
                          unsigned short* __restrict__ zlo) {
    int bh = blockIdx.x;
    int b = bh >> 5, h = bh & 31;
    __shared__ unsigned short thi[32 * 256];
    __shared__ unsigned short tlo[32 * 256];
    int t = threadIdx.x;
    int w = t & 31, c0 = t >> 5;
    for (int i = 0; i < 32; ++i) {
        int c = i * 8 + c0;
        float v = z[(size_t)b * 262144 + (size_t)c * 1024 + h * 32 + w];
        unsigned short hb = f2bf(v);
        thi[w * 256 + c] = hb;
        tlo[w * 256 + c] = f2bf(v - bf2f(hb));
    }
    __syncthreads();
    for (int wo = 0; wo < 32; ++wo) {
        size_t n = (size_t)b * 1024 + h * 32 + wo;
        zhi[n * DIM + t] = thi[wo * 256 + t];
        zlo[n * DIM + t] = tlo[wo * 256 + t];
    }
}

// --------------------------------------------------------------------------
// Main GEMM + argmin.  4 waves/block, 16 rows per wave (64 rows per block),
// 256 workgroups.  Each 16-code column tile (hi+lo, 16 KB) is staged into
// LDS with double buffering via async global->LDS (ASYNCcnt).
// s[n][k] = ||e_k||^2 - 2 * z_n . e_k ; argmin_k.
// Dot product via 3-term bf16 WMMA: hi*hi + hi*lo + lo*hi (~fp32 accurate).
// --------------------------------------------------------------------------
__global__ void __launch_bounds__(128)
vq_gemm_argmin(const unsigned short* __restrict__ zhi_,
               const unsigned short* __restrict__ zlo_,
               const unsigned short* __restrict__ ehi_,
               const unsigned short* __restrict__ elo_,
               const float* __restrict__ enorm,
               int* __restrict__ idxout) {
    // [buffer][hi/lo][16 codes x 256 dims]
    __shared__ __attribute__((aligned(32))) unsigned short ldsB[2][2][4096];

    const int tid  = threadIdx.x;
    const int wave = tid >> 5;
    const int lane = tid & 31;
    const int m    = lane & 15;     // row within 16-row tile (A), col (B/C)
    const int half = lane >> 4;     // lane half selects K-subrange / M-half
    const int rowbase = blockIdx.x * 64 + wave * 16;

    const __bf16* zh = (const __bf16*)zhi_;
    const __bf16* zl = (const __bf16*)zlo_;

    // Preload A fragments: 8 K-slices of 16x32 bf16, hi and lo.
    // A k order used here (same as B): lane<16 holds row m, K 0..15 contiguous;
    // lane>=16 holds row m, K 16..31 contiguous.
    v16bf a_hi[8], a_lo[8];
    const size_t abase = (size_t)(rowbase + m) * DIM;
    #pragma unroll
    for (int s = 0; s < 8; ++s) {
        int kb = s * 32;
        v8bf h0 = *(const v8bf*)(zh + abase + kb + half * 16);
        v8bf h1 = *(const v8bf*)(zh + abase + kb + 8 + half * 16);
        a_hi[s] = __builtin_shufflevector(h0, h1, 0, 1, 2, 3, 4, 5, 6, 7,
                                          8, 9, 10, 11, 12, 13, 14, 15);
        v8bf l0 = *(const v8bf*)(zl + abase + kb + half * 16);
        v8bf l1 = *(const v8bf*)(zl + abase + kb + 8 + half * 16);
        a_lo[s] = __builtin_shufflevector(l0, l1, 0, 1, 2, 3, 4, 5, 6, 7,
                                          8, 9, 10, 11, 12, 13, 14, 15);
    }

    float best[8];
    int   bidx[8];
    #pragma unroll
    for (int r = 0; r < 8; ++r) { best[r] = 3.4e38f; bidx[r] = 0; }

    const int col = m;  // C/D: lane&15 is the column

    // ---- stage helper (1024 16B units per tile, 8 per thread) ----
    auto stage = [&](int ct, int buf) {
        #pragma unroll
        for (int i = 0; i < 8; ++i) {
            int u = tid + i * 128;
            int which = u >> 9;            // 0 = hi, 1 = lo
            int off   = (u & 511) * 8;     // ushort offset
            const unsigned short* src = (which ? elo_ : ehi_) +
                                        (size_t)ct * (16 * DIM) + off;
            unsigned short* dst = &ldsB[buf][which][off];
#if HAVE_ASYNC_LDS
            __builtin_amdgcn_global_load_async_to_lds_b128(
                (GLOBAL_AS v4i*)(uintptr_t)src,
                (LDS_AS v4i*)(uintptr_t)dst, 0, 0);
#else
            *(uint4*)dst = *(const uint4*)src;
#endif
        }
    };

    // ---- double-buffered pipeline over 512 column tiles ----
    stage(0, 0);
    WAIT_ASYNC();
    __syncthreads();

    for (int ct = 0; ct < NCODES / 16; ++ct) {
        const int cur = ct & 1;
        if (ct + 1 < NCODES / 16) stage(ct + 1, cur ^ 1);

        v8f acc = {};
        #pragma unroll
        for (int s = 0; s < 8; ++s) {
            // B k order used here (same as A): lane<16 -> column N=lane, K 0..15
            // contiguous; lane>=16 -> column N=lane-16, K 16..31.
            const __bf16* bh =
                (const __bf16*)&ldsB[cur][0][col * DIM + s * 32 + half * 16];
            const __bf16* bl =
                (const __bf16*)&ldsB[cur][1][col * DIM + s * 32 + half * 16];
            v16bf Bh = *(const v16bf*)bh;
            v16bf Bl = *(const v16bf*)bl;
            acc = __builtin_amdgcn_wmma_f32_16x16x32_bf16(
                false, a_hi[s], false, Bh, (short)0, acc, false, false);
            acc = __builtin_amdgcn_wmma_f32_16x16x32_bf16(
                false, a_hi[s], false, Bl, (short)0, acc, false, false);
            acc = __builtin_amdgcn_wmma_f32_16x16x32_bf16(
                false, a_lo[s], false, Bh, (short)0, acc, false, false);
        }

        int colg = ct * 16 + col;
        float nrm = enorm[colg];
        #pragma unroll
        for (int r = 0; r < 8; ++r) {
            // C layout: VGPR r holds row (r + half*8), col = lane&15
            float sv = nrm - 2.0f * acc[r];
            bool take = sv < best[r];
            best[r] = take ? sv : best[r];
            bidx[r] = take ? colg : bidx[r];
        }

        // next tile's async copies must land; everyone done reading cur
        WAIT_ASYNC();
        __syncthreads();
    }

    // Cross-lane argmin within each 16-lane half (rows differ between halves).
    #pragma unroll
    for (int r = 0; r < 8; ++r) {
        float v = best[r];
        int   ix = bidx[r];
        #pragma unroll
        for (int off = 8; off >= 1; off >>= 1) {
            float ov = __shfl_xor(v, off, 32);
            int   oi = __shfl_xor(ix, off, 32);
            bool take = (ov < v) || (ov == v && oi < ix);  // first-index ties
            v  = take ? ov : v;
            ix = take ? oi : ix;
        }
        if (col == 0) idxout[rowbase + half * 8 + r] = ix;
    }
}

// --------------------------------------------------------------------------
// z_q gather (exact fp32 emb values) + element-wise commitment loss.
// --------------------------------------------------------------------------
__global__ void vq_gather_loss(const float* __restrict__ z,
                               const float* __restrict__ emb,
                               const int* __restrict__ idx,
                               float* __restrict__ out,
                               float* __restrict__ lacc) {
    int t = threadIdx.x;
    size_t flat = (size_t)blockIdx.x * 256 + t;
    int w = (int)(flat & 31);
    int h = (int)((flat >> 5) & 31);
    int c = (int)((flat >> 10) & 255);
    int b = (int)(flat >> 18);
    int n = b * 1024 + h * 32 + w;
    int k = idx[n];
    float e  = emb[(size_t)k * DIM + c];
    float zc = z[flat];
    out[flat] = e;  // straight-through forward value == z_q
    float d = e - zc;
    __shared__ float red[256];
    red[t] = d * d;
    __syncthreads();
    for (int s = 128; s > 0; s >>= 1) {
        if (t < s) red[t] += red[t + s];
        __syncthreads();
    }
    if (t == 0) atomicAdd(lacc, red[0]);
}

// --------------------------------------------------------------------------
// indices-as-float, one-hot scatter (region pre-zeroed), histogram.
// --------------------------------------------------------------------------
__global__ void vq_onehot(const int* __restrict__ idx,
                          float* __restrict__ out,
                          int* __restrict__ hist) {
    int n = blockIdx.x * 256 + threadIdx.x;
    int k = idx[n];
    out[OUT_IDX + (size_t)n] = (float)k;
    atomicAdd(&hist[k], 1);
    out[OUT_ENC + (size_t)n * NCODES + k] = 1.0f;
}

// --------------------------------------------------------------------------
// loss scalar + perplexity from histogram.  Single block.
// --------------------------------------------------------------------------
__global__ void vq_finalize(const int* __restrict__ hist,
                            const float* __restrict__ lacc,
                            float* __restrict__ out) {
    int t = threadIdx.x;
    __shared__ float red[256];
    float s = 0.0f;
    for (int k = t; k < NCODES; k += 256) {
        float p = (float)hist[k] * (1.0f / (float)NVEC);
        s += p * logf(p + 1e-10f);
    }
    red[t] = s;
    __syncthreads();
    for (int st = 128; st > 0; st >>= 1) {
        if (t < st) red[t] += red[t + st];
        __syncthreads();
    }
    if (t == 0) {
        out[OUT_LOSS] = 0.25f * lacc[0] / (float)NELEM;  // BETA = 0.25
        out[OUT_PERP] = expf(-red[0]);
    }
}

// --------------------------------------------------------------------------
extern "C" void kernel_launch(void* const* d_in, const int* in_sizes, int n_in,
                              void* d_out, int out_size, void* d_ws,
                              size_t ws_size, hipStream_t stream) {
    const float* z   = (const float*)d_in[0];
    const float* emb = (const float*)d_in[1];
    float* out = (float*)d_out;
    char*  ws  = (char*)d_ws;

    unsigned short* emb_hi = (unsigned short*)(ws + WS_EMB_HI);
    unsigned short* emb_lo = (unsigned short*)(ws + WS_EMB_LO);
    unsigned short* z_hi   = (unsigned short*)(ws + WS_Z_HI);
    unsigned short* z_lo   = (unsigned short*)(ws + WS_Z_LO);
    float* enorm = (float*)(ws + WS_ENORM);
    int*   idx   = (int*)(ws + WS_IDX);
    int*   hist  = (int*)(ws + WS_HIST);
    float* lacc  = (float*)(ws + WS_LACC);

    // zero histogram + loss accumulator (contiguous), and the one-hot region
    (void)hipMemsetAsync(hist, 0, NCODES * sizeof(int) + sizeof(float), stream);
    (void)hipMemsetAsync(out + OUT_ENC, 0,
                         (size_t)NVEC * NCODES * sizeof(float), stream);

    vq_emb_prep<<<NCODES, 256, 0, stream>>>(emb, emb_hi, emb_lo, enorm);
    vq_z_prep<<<512, 256, 0, stream>>>(z, z_hi, z_lo);
    vq_gemm_argmin<<<NVEC / 64, 128, 0, stream>>>(z_hi, z_lo, emb_hi, emb_lo,
                                                  enorm, idx);
    vq_gather_loss<<<NELEM / 256, 256, 0, stream>>>(z, emb, idx, out, lacc);
    vq_onehot<<<NVEC / 256, 256, 0, stream>>>(idx, out, hist);
    vq_finalize<<<1, 256, 0, stream>>>(hist, lacc, out);
}
